// PeriodicNetwork_Ver1_54889682043499
// MI455X (gfx1250) — compile-verified, matrix operand order checked by reading, not run on hardware
//
#include <hip/hip_runtime.h>

// ---------------------------------------------------------------------------
// Periodic e3nn-style GNN forward on MI455X (gfx1250, wave32, WMMA bf16).
// Edge GEMM: 32-edge tiles, sh-channel scaling in registers (v_pk_mul_bf16),
// B fragments reused across two M-tiles (2 wmma / B-frag), and the channel
// loop software-pipelined (rotation) so the WMMA->VALU WAR hazard slots are
// filled with the next channel's pk_mul/loads instead of v_nop.
// ---------------------------------------------------------------------------

#define N_NODES  20000
#define N_EDGES  240000
#define N_GRAPHS 200
#define D_IN     118
#define H        256
#define NB       10
#define RH       100
#define LAYERS   3
#define SHC      9
#define MAX_RADIUS 5.0f
#define INV_SQRT_DEG 0.28867513459481287f   // 1/sqrt(12)

typedef __bf16 bf16;
typedef __attribute__((ext_vector_type(16))) __bf16 v16bf;
typedef __attribute__((ext_vector_type(8)))  __bf16 v8bf;
typedef __attribute__((ext_vector_type(8)))  float  v8f;

__device__ __forceinline__ v16bf load_frag(const bf16* p0, const bf16* p1) {
    union { v16bf v; v8bf h[2]; } u;
    u.h[0] = *(const v8bf*)p0;   // K = base .. base+7
    u.h[1] = *(const v8bf*)p1;   // K = base+16 .. base+23
    return u.v;
}

__device__ __forceinline__ float gelu_tanh(float x) {
    const float k0 = 0.7978845608028654f;   // sqrt(2/pi)
    float inner = k0 * (x + 0.044715f * x * x * x);
    return 0.5f * x * (1.0f + tanhf(inner));
}

// ---------------------------------------------------------------------------
// Prep: transpose + bf16-convert tensor-product weights.  Wtp (L,SH,H,H)
// row-major (k-major) -> wtp_t[(la*H + n)*H + k]  (n-major, contiguous k).
// ---------------------------------------------------------------------------
__global__ void prep_wtp_kernel(const float* __restrict__ Wtp, bf16* __restrict__ wtp_t) {
    int idx = blockIdx.x * 256 + threadIdx.x;
    if (idx >= LAYERS * SHC * H * H) return;
    int k  = idx & (H - 1);
    int n  = (idx >> 8) & (H - 1);
    int la = idx >> 16;
    wtp_t[((size_t)la * H + n) * H + k] = (bf16)Wtp[((size_t)la * H + k) * H + n];
}

// Wself/Wskip (L,H,H) -> concatenated K=512, n-major bf16
__global__ void prep_wns_kernel(const float* __restrict__ Wself,
                                const float* __restrict__ Wskip,
                                bf16* __restrict__ wns_t) {
    int idx = blockIdx.x * 256 + threadIdx.x;
    if (idx >= LAYERS * H * 2 * H) return;
    int k = idx & (2 * H - 1);
    int n = (idx >> 9) & (H - 1);
    int l = idx >> 17;
    float v = (k < H) ? Wself[((size_t)l * H + k) * H + n]
                      : Wskip[((size_t)l * H + (k - H)) * H + n];
    wns_t[((size_t)l * H + n) * (2 * H) + k] = (bf16)v;
}

// ---------------------------------------------------------------------------
// Edge geometry: edge_vec, length, unit vec, sph-harm (l<=2), cosine basis.
// ---------------------------------------------------------------------------
__global__ void edge_geom_kernel(const float* __restrict__ pos,
                                 const float* __restrict__ shift,
                                 const float* __restrict__ lattice,
                                 const int*   __restrict__ ei,
                                 const int*   __restrict__ batch,
                                 float* __restrict__ sh_g,
                                 float* __restrict__ emb_g) {
    int e = blockIdx.x * 256 + threadIdx.x;
    if (e >= N_EDGES) return;
    int s = ei[e], d = ei[N_EDGES + e];
    int g = batch[s];
    float sx = shift[(size_t)e * 3], sy = shift[(size_t)e * 3 + 1], sz = shift[(size_t)e * 3 + 2];
    const float* Lm = lattice + (size_t)g * 9;
    float vx = pos[(size_t)d * 3]     - pos[(size_t)s * 3]     + sx * Lm[0] + sy * Lm[3] + sz * Lm[6];
    float vy = pos[(size_t)d * 3 + 1] - pos[(size_t)s * 3 + 1] + sx * Lm[1] + sy * Lm[4] + sz * Lm[7];
    float vz = pos[(size_t)d * 3 + 2] - pos[(size_t)s * 3 + 2] + sx * Lm[2] + sy * Lm[5] + sz * Lm[8];
    float len = sqrtf(vx * vx + vy * vy + vz * vz);
    float inv = 1.0f / (len + 1e-12f);
    float x = vx * inv, y = vy * inv, z = vz * inv;
    const float c1 = 1.7320508075688772f;   // sqrt(3)
    const float c2 = 3.8729833462074170f;   // sqrt(15)
    const float c3 = 1.1180339887498949f;   // sqrt(5)/2
    float* o = sh_g + (size_t)e * SHC;
    o[0] = 1.0f;
    o[1] = c1 * x;  o[2] = c1 * y;  o[3] = c1 * z;
    o[4] = c2 * x * y;  o[5] = c2 * y * z;
    o[6] = c3 * (3.0f * z * z - 1.0f);
    o[7] = c2 * x * z;  o[8] = 0.5f * c2 * (x * x - y * y);
    const float step = MAX_RADIUS / (float)(NB + 1);
    const float sq   = 3.1622776601683795f; // sqrt(10)
    float* eo = emb_g + (size_t)e * NB;
    #pragma unroll
    for (int b = 0; b < NB; ++b) {
        float diff = (len - (float)(b + 1) * step) / step;
        eo[b] = (fabsf(diff) < 1.0f) ? __cosf(1.5707963267948966f * diff) * sq : 0.0f;
    }
}

// ---------------------------------------------------------------------------
// Embedding: h = x @ W_embed (K=118, coalesced f32; ~1.2 GFLOP).
// ---------------------------------------------------------------------------
__global__ __launch_bounds__(256) void embed_kernel(const float* __restrict__ x,
                                                    const float* __restrict__ W,
                                                    float* __restrict__ h_f,
                                                    bf16*  __restrict__ h_bf) {
    int node = blockIdx.x;
    int n = threadIdx.x;
    const float* xr = x + (size_t)node * D_IN;
    float acc = 0.0f;
    for (int k = 0; k < D_IN; ++k)
        acc += xr[k] * W[(size_t)k * H + n];
    h_f[(size_t)node * H + n] = acc;
    h_bf[(size_t)node * H + n] = (bf16)acc;
}

__global__ void zero_kernel(float* __restrict__ p, int n) {
    int i = blockIdx.x * 256 + threadIdx.x;
    if (i < n) p[i] = 0.0f;
}

// ---------------------------------------------------------------------------
// Fused edge message kernel (per layer).  Block = 32 edges, 8 waves.
//   radial = silu(emb@r1+b1)@r2
//   hs     = h_bf16[src] * radial  -> bf16 in LDS (sh NOT applied)
//   GEMM:  per wave, 2 M-tiles x 2 N-tiles; sh_a applied as per-lane
//          register splat; channel loop software-pipelined so next-channel
//          pk_muls/loads fill the WMMA->VALU WAR hazard slots.
//   atomicAdd into agg[dst] * 1/sqrt(12)
// ---------------------------------------------------------------------------
__global__ __launch_bounds__(256) void edge_msg_kernel(
    const bf16*  __restrict__ h_bf,
    const float* __restrict__ sh_g,
    const float* __restrict__ emb_g,
    const int*   __restrict__ ei,
    const float* __restrict__ r1,
    const float* __restrict__ b1,
    const float* __restrict__ r2,
    const bf16*  __restrict__ wtp_t,
    float* __restrict__ agg,
    int layer)
{
    __shared__ float emb_s[32][NB];
    __shared__ float sh_s[32][SHC];
    __shared__ int   src_s[32];
    __shared__ int   dst_s[32];
    __shared__ float rh_s[32][RH];
    __shared__ bf16  hs_bf[32][H + 8];   // 264*2B rows -> conflict-free b128 frags

    const int tid = threadIdx.x;
    const int e0  = blockIdx.x * 32;

    if (tid < 32) {
        src_s[tid] = ei[e0 + tid];
        dst_s[tid] = ei[N_EDGES + e0 + tid];
    }
    for (int idx = tid; idx < 32 * NB; idx += 256)
        emb_s[idx / NB][idx % NB] = emb_g[(size_t)(e0 + idx / NB) * NB + idx % NB];
    for (int idx = tid; idx < 32 * SHC; idx += 256)
        sh_s[idx / SHC][idx % SHC] = sh_g[(size_t)(e0 + idx / SHC) * SHC + idx % SHC];
    __syncthreads();

    // radial hidden: silu(emb @ r1 + b1)   (32 x 100)
    for (int idx = tid; idx < 32 * RH; idx += 256) {
        int i = idx / RH, j = idx % RH;
        float v = b1[layer * RH + j];
        #pragma unroll
        for (int t = 0; t < NB; ++t)
            v += emb_s[i][t] * r1[((size_t)layer * NB + t) * RH + j];
        rh_s[i][j] = v / (1.0f + __expf(-v));
    }
    __syncthreads();

    // hs = h[src] * (rh @ r2)  -> bf16 LDS (32 x 256)
    for (int idx = tid; idx < 32 * H; idx += 256) {
        int i = idx >> 8, k = idx & (H - 1);
        float r = 0.0f;
        for (int j = 0; j < RH; ++j)
            r += rh_s[i][j] * r2[((size_t)layer * RH + j) * H + k];
        hs_bf[i][k] = (bf16)((float)h_bf[(size_t)src_s[i] * H + k] * r);
    }
    __syncthreads();

    const int lane  = tid & 31;
    const int wave  = tid >> 5;
    const int m     = lane & 15;   // row within M-tile (A) / column (B)
    const int hi    = lane >> 4;   // K half select
    const int ncol0 = wave * 32;   // 32 output columns per wave (2 N-tiles)

    // per-lane sph-harm scalars (row m for M-tile 0, row 16+m for M-tile 1)
    bf16 sh0b[SHC], sh1b[SHC];
    #pragma unroll
    for (int a = 0; a < SHC; ++a) {
        sh0b[a] = (bf16)sh_s[m][a];
        sh1b[a] = (bf16)sh_s[16 + m][a];
    }

    v8f acc00 = {}, acc01 = {}, acc10 = {}, acc11 = {};
    const bf16* wl = wtp_t + (size_t)layer * SHC * H * H;

    for (int kc = 0; kc < 8; ++kc) {
        int kb = kc * 32 + hi * 8;
        v16bf a0 = load_frag(&hs_bf[m][kb],      &hs_bf[m][kb + 16]);
        v16bf a1 = load_frag(&hs_bf[16 + m][kb], &hs_bf[16 + m][kb + 16]);
        const bf16* bp0 = wl + (size_t)(ncol0 + m) * H + kb;
        const bf16* bp1 = wl + (size_t)(ncol0 + 16 + m) * H + kb;

        // pipeline prologue: channel 0 operands
        v16bf bv0 = load_frag(bp0, bp0 + 16);
        v16bf bv1 = load_frag(bp1, bp1 + 16);
        v16bf c0  = a0 * sh0b[0];
        v16bf c1  = a1 * sh1b[0];

        #pragma unroll
        for (int a = 0; a < SHC; ++a) {
            // issue next channel's loads + scaling BEFORE this channel's WMMAs
            v16bf nb0 = bv0, nb1 = bv1, n0 = c0, n1 = c1;
            if (a + 1 < SHC) {
                const bf16* q0 = bp0 + (size_t)(a + 1) * H * H;
                const bf16* q1 = bp1 + (size_t)(a + 1) * H * H;
                nb0 = load_frag(q0, q0 + 16);
                nb1 = load_frag(q1, q1 + 16);
                n0  = a0 * sh0b[a + 1];
                n1  = a1 * sh1b[a + 1];
            }
            acc00 = __builtin_amdgcn_wmma_f32_16x16x32_bf16(
                false, c0, false, bv0, (short)0, acc00, false, false);
            acc01 = __builtin_amdgcn_wmma_f32_16x16x32_bf16(
                false, c0, false, bv1, (short)0, acc01, false, false);
            acc10 = __builtin_amdgcn_wmma_f32_16x16x32_bf16(
                false, c1, false, bv0, (short)0, acc10, false, false);
            acc11 = __builtin_amdgcn_wmma_f32_16x16x32_bf16(
                false, c1, false, bv1, (short)0, acc11, false, false);
            bv0 = nb0; bv1 = nb1; c0 = n0; c1 = n1;
        }
    }

    // scatter-add (segment_sum over edge_dst) with 1/sqrt(NUM_NEIGHBORS)
    #pragma unroll
    for (int v = 0; v < 8; ++v) {
        int r0 = v + 8 * hi;          // M-tile 0 rows
        int r1i = 16 + r0;            // M-tile 1 rows
        int d0 = dst_s[r0];
        int d1 = dst_s[r1i];
        atomicAdd(&agg[(size_t)d0 * H + ncol0 + m],      acc00[v] * INV_SQRT_DEG);
        atomicAdd(&agg[(size_t)d0 * H + ncol0 + 16 + m], acc01[v] * INV_SQRT_DEG);
        atomicAdd(&agg[(size_t)d1 * H + ncol0 + m],      acc10[v] * INV_SQRT_DEG);
        atomicAdd(&agg[(size_t)d1 * H + ncol0 + 16 + m], acc11[v] * INV_SQRT_DEG);
    }
}

// ---------------------------------------------------------------------------
// Node update (per layer): h = gelu([agg | h] @ [Wself; Wskip])  (K=512 WMMA)
// 32-row tiles; B fragments reused across two M-tiles; loads for kc+1 issued
// ahead of kc's WMMAs.  In-place safe (each block owns its 32 rows).
// ---------------------------------------------------------------------------
__global__ __launch_bounds__(256) void node_update_kernel(
    const float* __restrict__ agg,
    float* __restrict__ h_f,
    bf16*  __restrict__ h_bf,
    const bf16* __restrict__ wns_t,
    int layer)
{
    __shared__ bf16 As[32][2 * H + 8];   // 520*2B rows, 16B-aligned frags

    const int tid = threadIdx.x;
    const int n0  = blockIdx.x * 32;

    for (int idx = tid; idx < 32 * 2 * H; idx += 256) {
        int i = idx >> 9, k = idx & (2 * H - 1);
        float v = (k < H) ? agg[(size_t)(n0 + i) * H + k]
                          : h_f[(size_t)(n0 + i) * H + (k - H)];
        As[i][k] = (bf16)v;
    }
    __syncthreads();

    const int lane  = tid & 31;
    const int wave  = tid >> 5;
    const int m     = lane & 15;
    const int hi    = lane >> 4;
    const int ncol0 = wave * 32;

    v8f acc00 = {}, acc01 = {}, acc10 = {}, acc11 = {};
    const bf16* bb0 = wns_t + (size_t)layer * H * 2 * H + (size_t)(ncol0 + m) * (2 * H);
    const bf16* bb1 = bb0 + (size_t)16 * (2 * H);

    // prologue (kc = 0)
    int kb = hi * 8;
    v16bf a0  = load_frag(&As[m][kb],      &As[m][kb + 16]);
    v16bf a1  = load_frag(&As[16 + m][kb], &As[16 + m][kb + 16]);
    v16bf bv0 = load_frag(bb0 + kb, bb0 + kb + 16);
    v16bf bv1 = load_frag(bb1 + kb, bb1 + kb + 16);

    for (int kc = 0; kc < 16; ++kc) {
        v16bf na0 = a0, na1 = a1, nb0 = bv0, nb1 = bv1;
        if (kc + 1 < 16) {
            int nkb = (kc + 1) * 32 + hi * 8;
            na0 = load_frag(&As[m][nkb],      &As[m][nkb + 16]);
            na1 = load_frag(&As[16 + m][nkb], &As[16 + m][nkb + 16]);
            nb0 = load_frag(bb0 + nkb, bb0 + nkb + 16);
            nb1 = load_frag(bb1 + nkb, bb1 + nkb + 16);
        }
        acc00 = __builtin_amdgcn_wmma_f32_16x16x32_bf16(
            false, a0, false, bv0, (short)0, acc00, false, false);
        acc01 = __builtin_amdgcn_wmma_f32_16x16x32_bf16(
            false, a0, false, bv1, (short)0, acc01, false, false);
        acc10 = __builtin_amdgcn_wmma_f32_16x16x32_bf16(
            false, a1, false, bv0, (short)0, acc10, false, false);
        acc11 = __builtin_amdgcn_wmma_f32_16x16x32_bf16(
            false, a1, false, bv1, (short)0, acc11, false, false);
        a0 = na0; a1 = na1; bv0 = nb0; bv1 = nb1;
    }

    #pragma unroll
    for (int v = 0; v < 8; ++v) {
        int row0 = n0 + v + 8 * hi;
        int row1 = row0 + 16;
        float g00 = gelu_tanh(acc00[v]);
        float g01 = gelu_tanh(acc01[v]);
        float g10 = gelu_tanh(acc10[v]);
        float g11 = gelu_tanh(acc11[v]);
        h_f[(size_t)row0 * H + ncol0 + m]       = g00;
        h_f[(size_t)row0 * H + ncol0 + 16 + m]  = g01;
        h_f[(size_t)row1 * H + ncol0 + m]       = g10;
        h_f[(size_t)row1 * H + ncol0 + 16 + m]  = g11;
        h_bf[(size_t)row0 * H + ncol0 + m]      = (bf16)g00;
        h_bf[(size_t)row0 * H + ncol0 + 16 + m] = (bf16)g01;
        h_bf[(size_t)row1 * H + ncol0 + m]      = (bf16)g10;
        h_bf[(size_t)row1 * H + ncol0 + 16 + m] = (bf16)g11;
    }
}

// ---------------------------------------------------------------------------
// Readout: node_out = h @ Wout; segment mean per graph.
// sums layout: [0..199]=sums, [200..399]=counts
// ---------------------------------------------------------------------------
__global__ void readout_kernel(const float* __restrict__ h_f,
                               const float* __restrict__ Wout,
                               const int*   __restrict__ batch,
                               float* __restrict__ sums) {
    int n = blockIdx.x * 256 + threadIdx.x;
    if (n >= N_NODES) return;
    const float* hr = h_f + (size_t)n * H;
    float acc = 0.0f;
    for (int k = 0; k < H; ++k) acc += hr[k] * Wout[k];
    int g = batch[n];
    atomicAdd(&sums[g], acc);
    atomicAdd(&sums[N_GRAPHS + g], 1.0f);
}

__global__ void mean_kernel(const float* __restrict__ sums, float* __restrict__ out) {
    int g = threadIdx.x;
    if (g < N_GRAPHS) out[g] = sums[g] / fmaxf(sums[N_GRAPHS + g], 1.0f);
}

// ---------------------------------------------------------------------------
extern "C" void kernel_launch(void* const* d_in, const int* in_sizes, int n_in,
                              void* d_out, int out_size, void* d_ws, size_t ws_size,
                              hipStream_t stream) {
    (void)in_sizes; (void)n_in; (void)out_size; (void)ws_size;

    const float* x       = (const float*)d_in[0];
    const float* pos     = (const float*)d_in[1];
    const float* shift   = (const float*)d_in[2];
    const float* lattice = (const float*)d_in[3];
    const float* W_embed = (const float*)d_in[4];
    const float* r1      = (const float*)d_in[5];
    const float* b1      = (const float*)d_in[6];
    const float* r2      = (const float*)d_in[7];
    const float* Wtp     = (const float*)d_in[8];
    const float* Wself   = (const float*)d_in[9];
    const float* Wskip   = (const float*)d_in[10];
    const float* Wout    = (const float*)d_in[11];
    const int*   ei      = (const int*)d_in[12];
    const int*   batch   = (const int*)d_in[13];
    float* out = (float*)d_out;

    char* w = (char*)d_ws;
    auto alloc = [&](size_t bytes) -> char* {
        char* p = w; w += (bytes + 255) & ~(size_t)255; return p;
    };
    float* h_f   = (float*)alloc((size_t)N_NODES * H * 4);
    bf16*  h_bf  = (bf16*) alloc((size_t)N_NODES * H * 2);
    float* agg   = (float*)alloc((size_t)N_NODES * H * 4);
    float* sh_g  = (float*)alloc((size_t)N_EDGES * SHC * 4);
    float* emb_g = (float*)alloc((size_t)N_EDGES * NB * 4);
    bf16*  wtp_t = (bf16*) alloc((size_t)LAYERS * SHC * H * H * 2);
    bf16*  wns_t = (bf16*) alloc((size_t)LAYERS * H * 2 * H * 2);
    float* sums  = (float*)alloc((size_t)2 * N_GRAPHS * 4);

    prep_wtp_kernel<<<(LAYERS * SHC * H * H + 255) / 256, 256, 0, stream>>>(Wtp, wtp_t);
    prep_wns_kernel<<<(LAYERS * H * 2 * H + 255) / 256, 256, 0, stream>>>(Wself, Wskip, wns_t);
    edge_geom_kernel<<<(N_EDGES + 255) / 256, 256, 0, stream>>>(pos, shift, lattice, ei, batch, sh_g, emb_g);
    embed_kernel<<<N_NODES, 256, 0, stream>>>(x, W_embed, h_f, h_bf);

    for (int l = 0; l < LAYERS; ++l) {
        zero_kernel<<<(N_NODES * H + 255) / 256, 256, 0, stream>>>(agg, N_NODES * H);
        edge_msg_kernel<<<N_EDGES / 32, 256, 0, stream>>>(h_bf, sh_g, emb_g, ei, r1, b1, r2, wtp_t, agg, l);
        node_update_kernel<<<N_NODES / 32, 256, 0, stream>>>(agg, h_f, h_bf, wns_t, l);
    }

    zero_kernel<<<1, 256, 0, stream>>>(sums, 2 * N_GRAPHS);
    readout_kernel<<<(N_NODES + 255) / 256, 256, 0, stream>>>(h_f, Wout, batch, sums);
    mean_kernel<<<1, 256, 0, stream>>>(sums, out);
}
